// STTFormer_72069551227132
// MI455X (gfx1250) — matrix operand order repaired and checked by reading, not compiled, not run on hardware
//
#include <hip/hip_runtime.h>
#include <hip/hip_bf16.h>
#include <stdint.h>

// ---------------------------------------------------------------------------
// CDNA5 (gfx1250) STTFormer forward.  Matrix work -> v_wmma_f32_16x16x32_bf16
// (wave32).  X tiles staged to LDS via the Tensor Data Mover when the tap
// shift is zero; masked manual staging otherwise.  f32 master activations +
// bf16 shadow copies written by fused GEMM epilogues.
// ---------------------------------------------------------------------------

typedef __bf16 bf16_t;
typedef __attribute__((ext_vector_type(16))) __bf16 v16bf;
typedef __attribute__((ext_vector_type(8)))  __bf16 v8bf;
typedef __attribute__((ext_vector_type(8)))  float  v8f;

#if defined(__HIP_DEVICE_COMPILE__) && __has_builtin(__builtin_amdgcn_tensor_load_to_lds)
#define HAVE_TDM 1
#else
#define HAVE_TDM 0
#endif

typedef unsigned int v4u_t __attribute__((ext_vector_type(4)));
typedef int          v8i_t __attribute__((ext_vector_type(8)));
typedef int          v4i_t __attribute__((ext_vector_type(4)));

#define BN_EPS 1e-5f

__device__ __forceinline__ float bf2f(bf16_t h) {
  unsigned short u = __builtin_bit_cast(unsigned short, h);
  unsigned int x = ((unsigned int)u) << 16;
  return __builtin_bit_cast(float, x);
}
__device__ __forceinline__ bf16_t f2bf(float f) {
  unsigned int x = __builtin_bit_cast(unsigned int, f);
  unsigned int r = x + 0x7fffu + ((x >> 16) & 1u);   // round-to-nearest-even
  unsigned short h = (unsigned short)(r >> 16);
  return __builtin_bit_cast(bf16_t, h);
}

// 16-bit A-matrix 16x32 lane->K map (ISA 7.12.2): lanes 0-15 hold K {0..7,16..23},
// lanes 16-31 hold K {8..15,24..31}.  (two contiguous 8-element runs per lane)
__device__ __forceinline__ int amapK(int i, int hf) { return (i & 7) + ((i >> 3) << 4) + hf * 8; }
// B-matrix 32x16: same K map as A: lanes 0-15 hold K {0..7,16..23}, lanes 16-31 hold K {8..15,24..31}.
__device__ __forceinline__ int bmapK(int i, int hf) { return (i & 7) + ((i >> 3) << 4) + hf * 8; }

// ---------------------------------------------------------------------------
// TDM: DMA a (tile_h x tile_w) bf16 tile of a row-major (rows x rowElems)
// tensor into LDS at lds_off, padding each 256B row with 16B so the LDS row
// stride is 136 bf16 elements.  Issued by one wave; EXEC is ignored by TDM.
// This toolchain exposes the 6-arg builtin: (v4u, v8i, v4i, v4i, v8i, cpol).
// ---------------------------------------------------------------------------
__device__ __forceinline__ void tdm_load_tile_bf16(const bf16_t* gsrc, unsigned lds_off,
                                                   unsigned rowElems, unsigned rows,
                                                   unsigned tile_w, unsigned tile_h)
{
#if HAVE_TDM
  unsigned long long ga = (unsigned long long)(uintptr_t)gsrc;
  v4u_t g0;
  g0[0] = 1u;                                          // count=1, user descriptor
  g0[1] = lds_off;                                     // lds_addr (bytes)
  g0[2] = (unsigned)(ga & 0xffffffffu);                // global_addr[31:0]
  g0[3] = ((unsigned)(ga >> 32) & 0x1ffffffu) | (2u << 30);   // addr[56:32] | type=2
  v8i_t g1;
  g1[0] = (int)((1u << 16)      // data_size = 1 -> 2 bytes
              | (1u << 20)      // pad_enable
              | (5u << 22)      // pad_interval: pad after 256B
              | (3u << 25));    // pad_amount: 4 DWORDs (16B)
  g1[1] = (int)((rowElems & 0xffffu) << 16);           // tensor_dim0[15:0]
  g1[2] = (int)((rowElems >> 16) | ((rows & 0xffffu) << 16));  // dim0 hi | dim1 lo
  g1[3] = (int)((rows >> 16) | (tile_w << 16));        // dim1 hi | tile_dim0
  g1[4] = (int)(tile_h & 0xffffu);                     // tile_dim1 (tile_dim2=0)
  g1[5] = (int)rowElems;                               // tensor_dim0_stride[31:0]
  g1[6] = 0;                                           // stride hi / dim1_stride lo
  g1[7] = 0;
  v4i_t gz = {};
  v8i_t gz8 = {};
  __builtin_amdgcn_tensor_load_to_lds(g0, g1, gz, gz, gz8, 0);
  __builtin_amdgcn_s_wait_tensorcnt(0);
#else
  (void)gsrc; (void)lds_off; (void)rowElems; (void)rows; (void)tile_w; (void)tile_h;
#endif
}

// ---------------------------------------------------------------------------
// Generic fused conv-as-GEMM:
//   Y[oc,p] = epilogue( sum_{tap, k<K0} W[oc,k,tap] * X[k, p + (tap-pad)*tapStride] )
// Block: 128 threads = 4 waves; tile 16 OC x 128 positions (2 acc per wave).
// ---------------------------------------------------------------------------
__global__ void __launch_bounds__(128)
stt_conv_gemm(const float* __restrict__ W, const bf16_t* __restrict__ X,
              const float* __restrict__ bias,
              const float* __restrict__ bn_g, const float* __restrict__ bn_b,
              const float* __restrict__ bn_m, const float* __restrict__ bn_v,
              const float* __restrict__ res,
              float* __restrict__ outF, bf16_t* __restrict__ outB,
              int OC, int K0, int P,
              int taps, int pad, int tapStride, int dimDiv, int dimSize,
              int do_lrelu)
{
  __shared__ bf16_t Wt[16 * 32];
  __shared__ bf16_t Xt[32 * 136];         // 128 cols + 8 pad (16B) per row

  const int tid  = threadIdx.x;
  const int lane = tid & 31;
  const int wave = tid >> 5;
  const int hf   = (lane >> 4) & 1;
  const int ln   = lane & 15;

  const int ocBase = blockIdx.y * 16;
  const int pBase  = blockIdx.x * 128;
  const int col0   = wave * 32 + ln;      // two 16-wide column sets per wave
  const int col1   = col0 + 16;

  const bf16_t bz = f2bf(0.0f);
  v8f acc0 = {}, acc1 = {};

  for (int tap = 0; tap < taps; ++tap) {
    const int d     = tap - pad;
    const int shift = d * tapStride;
    const bool useTdm = HAVE_TDM && (shift == 0);
    for (int kb = 0; kb < K0; kb += 32) {
      // ---- stage 16x32 weight tile (f32 -> bf16) ----
      for (int e = tid; e < 16 * 32; e += 128) {
        int m = e >> 5, kk = e & 31;
        int k = kb + kk;
        float w = (k < K0) ? W[((size_t)(ocBase + m) * K0 + k) * taps + tap] : 0.0f;
        Wt[m * 32 + kk] = f2bf(w);
      }
      // ---- stage 32x128 X tile ----
      if (useTdm) {
        if (wave == 0)
          tdm_load_tile_bf16(X + (size_t)kb * P + pBase,
                             (unsigned)(unsigned long long)(uintptr_t)(void*)Xt,
                             (unsigned)P, (unsigned)K0, 128u, 32u);
      } else {
        int row = tid >> 2;               // 0..31 (K)
        int seg = (tid & 3) * 32;         // 0/32/64/96
        int k   = kb + row;
        if (k + 32 < K0)
          __builtin_prefetch(&X[(size_t)(k + 32) * P + pBase + seg], 0, 1);
#pragma unroll
        for (int j = 0; j < 32; ++j) {
          int pp = seg + j;
          int p  = pBase + pp;
          int coord = (p / dimDiv) % dimSize + d;
          bool ok = (k < K0) && (coord >= 0) && (coord < dimSize);
          Xt[row * 136 + pp] = ok ? X[(size_t)k * P + (p + shift)] : bz;
        }
      }
      __syncthreads();
      // ---- fragments: A as two 16B LDS vector loads, B per-column ----
      v8bf alo = *(const v8bf*)&Wt[ln * 32 + hf * 8];
      v8bf ahi = *(const v8bf*)&Wt[ln * 32 + 16 + hf * 8];
      v16bf a = __builtin_shufflevector(alo, ahi, 0,1,2,3,4,5,6,7,8,9,10,11,12,13,14,15);
      v16bf b0, b1;
#pragma unroll
      for (int i = 0; i < 16; ++i) {
        int krow = bmapK(i, hf) * 136;
        b0[i] = Xt[krow + col0];
        b1[i] = Xt[krow + col1];
      }
      acc0 = __builtin_amdgcn_wmma_f32_16x16x32_bf16(false, a, false, b0,
                                                     (short)0, acc0, false, false);
      acc1 = __builtin_amdgcn_wmma_f32_16x16x32_bf16(false, a, false, b1,
                                                     (short)0, acc1, false, false);
      __syncthreads();
    }
  }

  // ---- fused epilogue: bias -> BN -> residual -> LeakyReLU -> f32 + bf16 ----
#pragma unroll
  for (int r = 0; r < 8; ++r) {
    int oc = ocBase + r + hf * 8;
    float bi = bias ? bias[oc] : 0.0f;
    float sc = 1.0f, sh = 0.0f;
    if (bn_g) {
      sc = bn_g[oc] * rsqrtf(bn_v[oc] + BN_EPS);
      sh = bn_b[oc] - bn_m[oc] * sc;
    }
    int p0 = pBase + col0, p1 = pBase + col1;
    float v0 = (acc0[r] + bi) * sc + sh;
    float v1 = (acc1[r] + bi) * sc + sh;
    if (res) { v0 += res[(size_t)oc * P + p0]; v1 += res[(size_t)oc * P + p1]; }
    if (do_lrelu) {
      v0 = (v0 >= 0.0f) ? v0 : 0.1f * v0;
      v1 = (v1 >= 0.0f) ? v1 : 0.1f * v1;
    }
    if (outF) { outF[(size_t)oc * P + p0] = v0; outF[(size_t)oc * P + p1] = v1; }
    if (outB) { outB[(size_t)oc * P + p0] = f2bf(v0); outB[(size_t)oc * P + p1] = f2bf(v1); }
  }
}

// ---------------------------------------------------------------------------
// Attention scores: att[n,h,u,v] = (sum_{c,t} q[n,h,c,t,u] k[n,h,c,t,v]) / (qd*T)
//                                  * values[h] + att_bias[h]
// ---------------------------------------------------------------------------
__global__ void __launch_bounds__(32)
stt_att_qk(const bf16_t* __restrict__ qkv, const float* __restrict__ values,
           const float* __restrict__ att_bias, float* __restrict__ att,
           int qd, int P, int T, int V, int Hn)
{
  const int lane = threadIdx.x;
  const int hf = lane >> 4, ln = lane & 15;
  const int ut = blockIdx.x, vt = blockIdx.y;
  const int nh = blockIdx.z;
  const int h = nh % Hn, n = nh / Hn;
  const int K = qd * T;
  const size_t pbase = (size_t)n * (T * V);
  const size_t qoff = (size_t)(h * qd) * P + pbase;
  const size_t koff = (size_t)((Hn + h) * qd) * P + pbase;
  const bf16_t bz = f2bf(0.0f);

  const int u = ut * 16 + ln;
  const int v = vt * 16 + ln;
  v8f acc = {};
  for (int kb = 0; kb < K; kb += 32) {
    v16bf a, b;
#pragma unroll
    for (int i = 0; i < 16; ++i) {
      int ka = kb + amapK(i, hf);
      bf16_t av = bz;
      if (ka < K && u < V) { int c = ka / T, t = ka % T; av = qkv[qoff + (size_t)c * P + t * V + u]; }
      a[i] = av;
      int kq = kb + bmapK(i, hf);
      bf16_t bv = bz;
      if (kq < K && v < V) { int c = kq / T, t = kq % T; bv = qkv[koff + (size_t)c * P + t * V + v]; }
      b[i] = bv;
    }
    acc = __builtin_amdgcn_wmma_f32_16x16x32_bf16(false, a, false, b,
                                                  (short)0, acc, false, false);
  }
  const float scale = values[h] / (float)(qd * T);
  const float ab = att_bias[h];
#pragma unroll
  for (int r = 0; r < 8; ++r) {
    int uu = ut * 16 + r + hf * 8;
    int vv = vt * 16 + ln;
    if (uu < V && vv < V)
      att[(((size_t)n * Hn + h) * V + uu) * V + vv] = acc[r] * scale + ab;
  }
}

// wave-per-row softmax (row length V=150), writes bf16 for the xs GEMM
__global__ void __launch_bounds__(32)
stt_softmax(const float* __restrict__ att, bf16_t* __restrict__ attb, int V)
{
  const int row = blockIdx.x;
  const int lane = threadIdx.x;
  const float* a = att + (size_t)row * V;
  float mx = -1e30f;
  for (int i = lane; i < V; i += 32) mx = fmaxf(mx, a[i]);
  for (int o = 16; o; o >>= 1) mx = fmaxf(mx, __shfl_xor(mx, o));
  float s = 0.0f;
  for (int i = lane; i < V; i += 32) s += __expf(a[i] - mx);
  for (int o = 16; o; o >>= 1) s += __shfl_xor(s, o);
  const float inv = 1.0f / s;
  for (int i = lane; i < V; i += 32)
    attb[(size_t)row * V + i] = f2bf(__expf(a[i] - mx) * inv);
}

// ---------------------------------------------------------------------------
// xs einsum: xs[n, h*ci+c, t, v] = sum_u x[c, n,t,u] * att[n,h,u,v]
// ---------------------------------------------------------------------------
__global__ void __launch_bounds__(32)
stt_xs(const bf16_t* __restrict__ xb, const bf16_t* __restrict__ attb,
       bf16_t* __restrict__ xs, int ci, int P, int T, int V, int Hn)
{
  const int lane = threadIdx.x;
  const int hf = lane >> 4, ln = lane & 15;
  const int vt = blockIdx.x, mt = blockIdx.y, nh = blockIdx.z;
  const int h = nh % Hn, n = nh / Hn;
  const bf16_t bz = f2bf(0.0f);

  const int m = mt * 16 + ln;               // (c,t) row for A fragment
  const int c = m / T, t = m % T;
  const size_t arow = (size_t)c * P + (size_t)n * (T * V) + (size_t)t * V;
  const bf16_t* attrow = attb + (size_t)nh * V * V;

  v8f acc = {};
  for (int kb = 0; kb < V; kb += 32) {
    v16bf a, b;
#pragma unroll
    for (int i = 0; i < 16; ++i) {
      int ka = kb + amapK(i, hf);
      a[i] = (ka < V) ? xb[arow + ka] : bz;
      int kq = kb + bmapK(i, hf);
      int v = vt * 16 + ln;
      b[i] = (kq < V && v < V) ? attrow[(size_t)kq * V + v] : bz;
    }
    acc = __builtin_amdgcn_wmma_f32_16x16x32_bf16(false, a, false, b,
                                                  (short)0, acc, false, false);
  }
#pragma unroll
  for (int r = 0; r < 8; ++r) {
    int mm = mt * 16 + r + hf * 8;
    int cc = mm / T, tt = mm % T;
    int vv = vt * 16 + ln;
    if (vv < V)
      xs[(size_t)(h * ci + cc) * P + (size_t)n * (T * V) + (size_t)tt * V + vv] = f2bf(acc[r]);
  }
}

// ---------------------------------------------------------------------------
// Small scalar kernels: input prep, 3->64 input conv, mean pool, FC head.
// ---------------------------------------------------------------------------
__global__ void stt_prep(const float* __restrict__ xin, float* __restrict__ x3, int P)
{
  int i = blockIdx.x * blockDim.x + threadIdx.x;
  if (i >= 3 * P) return;
  int c = i / P, p = i % P;
  int n = p / 3000, r = p % 3000;
  int t = r / 150, v150 = r % 150;
  int e = n / 32, b = n % 32;
  int t120 = t * 6 + v150 / 25, v25 = v150 % 25;
  x3[i] = xin[((((size_t)b * 3 + c) * 120 + t120) * 25 + v25) * 2 + e];
}

__global__ void stt_inconv(const float* __restrict__ x3, const float* __restrict__ w,
                           const float* __restrict__ b,
                           const float* __restrict__ g, const float* __restrict__ be,
                           const float* __restrict__ m, const float* __restrict__ va,
                           const float* __restrict__ pes,
                           float* __restrict__ xf, bf16_t* __restrict__ xb, int C, int P)
{
  int i = blockIdx.x * blockDim.x + threadIdx.x;
  if (i >= C * P) return;
  int oc = i / P, p = i % P;
  float s = b[oc];
  for (int ic = 0; ic < 3; ++ic) s += w[oc * 3 + ic] * x3[(size_t)ic * P + p];
  float sc = g[oc] * rsqrtf(va[oc] + BN_EPS);
  s = (s - m[oc]) * sc + be[oc];
  s = (s >= 0.0f) ? s : 0.1f * s;
  s += pes[(size_t)oc * 3000 + (p % 3000)];
  xf[i] = s;
  xb[i] = f2bf(s);
}

__global__ void __launch_bounds__(128)
stt_pool(const float* __restrict__ x, float* __restrict__ y, int C, int P, int TV)
{
  int wv = blockIdx.x * 4 + (threadIdx.x >> 5);
  int lane = threadIdx.x & 31;
  int rows = (P / TV) * C;
  if (wv >= rows) return;
  int n = wv / C, c = wv % C;
  const float* p = x + (size_t)c * P + (size_t)n * TV;
  float s = 0.0f;
  for (int i = lane; i < TV; i += 32) s += p[i];
  for (int o = 16; o; o >>= 1) s += __shfl_xor(s, o);
  if (lane == 0) y[n * C + c] = s / (float)TV;
}

__global__ void stt_fc(const float* __restrict__ y, const float* __restrict__ fw,
                       const float* __restrict__ fb, float* __restrict__ out,
                       int C, int B, int CLS)
{
  int i = blockIdx.x * blockDim.x + threadIdx.x;
  if (i >= B * CLS) return;
  int b = i / CLS, k = i % CLS;
  float s = fb[k];
  for (int c = 0; c < C; ++c)
    s += (y[b * C + c] + y[(b + B) * C + c]) * fw[k * C + c];
  out[i] = s;
}

// ---------------------------------------------------------------------------
// Host orchestration
// ---------------------------------------------------------------------------
extern "C" void kernel_launch(void* const* d_in, const int* in_sizes, int n_in,
                              void* d_out, int out_size, void* d_ws, size_t ws_size,
                              hipStream_t stream)
{
  (void)in_sizes; (void)n_in; (void)out_size; (void)ws_size;
  const int P = 192000, T = 20, V = 150, Hh = 3, Nn = 64, Cmax = 256;
  static const int CFGh[8][3] = {{64,64,16},{64,64,16},{64,128,32},{128,128,32},
                                 {128,256,64},{256,256,64},{256,256,64},{256,256,64}};

  // ---- walk the flattened parameter list (setup_inputs dict order) ----
  int ix = 0;
  auto F = [&]() -> const float* { return (const float*)d_in[ix++]; };
  struct BnP { const float *g, *b, *m, *v; };
  auto BNf = [&]() { BnP q; q.g = F(); q.b = F(); q.m = F(); q.v = F(); return q; };

  const float* x_in = F();
  const float* in_w = F(); const float* in_b = F();
  BnP in_bn = BNf();
  const float* pes  = F();
  const float* fc_w = F(); const float* fc_b = F();

  struct Blk {
    const float *qkv_w,*qkv_b,*values,*att_bias;
    const float *outs_w,*outs_b; BnP outs_bn;
    const float *ff_w,*ff_b;     BnP ff_bn;
    const float *outt_w,*outt_b; BnP outt_bn;
    const float *ress_w,*ress_b; BnP ress_bn;
    const float *rest_w,*rest_b; BnP rest_bn;
  };
  Blk blk[8];
  for (int i = 0; i < 8; ++i) {
    Blk& bp = blk[i];
    bp.qkv_w = F(); bp.qkv_b = F(); bp.values = F(); bp.att_bias = F();
    bp.outs_w = F(); bp.outs_b = F(); bp.outs_bn = BNf();
    bp.ff_w   = F(); bp.ff_b   = F(); bp.ff_bn   = BNf();
    bp.outt_w = F(); bp.outt_b = F(); bp.outt_bn = BNf();
    if (CFGh[i][0] != CFGh[i][1]) {
      bp.ress_w = F(); bp.ress_b = F(); bp.ress_bn = BNf();
      bp.rest_w = F(); bp.rest_b = F(); bp.rest_bn = BNf();
    } else {
      bp.ress_w = bp.ress_b = bp.rest_w = bp.rest_b = nullptr;
      bp.ress_bn = BnP{nullptr,nullptr,nullptr,nullptr};
      bp.rest_bn = BnP{nullptr,nullptr,nullptr,nullptr};
    }
  }

  // ---- workspace bump allocator ----
  char* wsp = (char*)d_ws; size_t off = 0;
  auto alloc = [&](size_t bytes) -> void* {
    void* p = wsp + off; off += (bytes + 255) & ~(size_t)255; return p;
  };
  float*  x3f   = (float*) alloc(sizeof(float) * 3 * (size_t)P);
  float*  xf    = (float*) alloc(sizeof(float) * Cmax * (size_t)P);
  bf16_t* xb    = (bf16_t*)alloc(2ull * Cmax * P);
  bf16_t* qkvb  = (bf16_t*)alloc(2ull * 384 * P);
  float*  attf  = (float*) alloc(4ull * Nn * Hh * V * V);
  bf16_t* attb  = (bf16_t*)alloc(2ull * Nn * Hh * V * V);
  bf16_t* xsb   = (bf16_t*)alloc(2ull * (3 * Cmax) * P);
  float*  xresf = (float*) alloc(4ull * Cmax * P);
  float*  y1f   = (float*) alloc(4ull * Cmax * P);
  bf16_t* y1b   = (bf16_t*)alloc(2ull * Cmax * P);
  float*  y2f   = (float*) alloc(4ull * Cmax * P);
  bf16_t* y2b   = (bf16_t*)alloc(2ull * Cmax * P);
  float*  restf = (float*) alloc(4ull * Cmax * P);
  float*  poolf = (float*) alloc(4ull * Nn * Cmax);

  // ---- input prep + input conv (+BN, lrelu, positional embedding) ----
  { int n = 3 * P;  stt_prep  <<<(n + 255) / 256, 256, 0, stream>>>(x_in, x3f, P); }
  { int n = 64 * P; stt_inconv<<<(n + 255) / 256, 256, 0, stream>>>(
      x3f, in_w, in_b, in_bn.g, in_bn.b, in_bn.m, in_bn.v, pes, xf, xb, 64, P); }

  // ---- 8 STA blocks ----
  for (int i = 0; i < 8; ++i) {
    const int ci = CFGh[i][0], co = CFGh[i][1], qd = CFGh[i][2];
    const Blk& bp = blk[i];
    const int OCq = 2 * Hh * qd;

    // qkv 1x1 conv -> bf16 (WMMA GEMM via TDM-staged tiles, bias only)
    stt_conv_gemm<<<dim3(P / 128, OCq / 16), 128, 0, stream>>>(
        bp.qkv_w, xb, bp.qkv_b, nullptr, nullptr, nullptr, nullptr,
        nullptr, nullptr, qkvb, OCq, ci, P, 1, 0, 1, 1, V, 0);

    // attention scores + softmax -> bf16
    stt_att_qk<<<dim3(10, 10, Nn * Hh), 32, 0, stream>>>(
        qkvb, bp.values, bp.att_bias, attf, qd, P, T, V, Hh);
    stt_softmax<<<Nn * Hh * V, 32, 0, stream>>>(attf, attb, V);

    // xs einsum -> bf16
    stt_xs<<<dim3(10, ci * T / 16, Nn * Hh), 32, 0, stream>>>(
        xb, attb, xsb, ci, P, T, V, Hh);

    // spatial residual
    const float* resPtr;
    if (ci != co) {
      stt_conv_gemm<<<dim3(P / 128, co / 16), 128, 0, stream>>>(
          bp.ress_w, xb, bp.ress_b, bp.ress_bn.g, bp.ress_bn.b, bp.ress_bn.m, bp.ress_bn.v,
          nullptr, xresf, nullptr, co, ci, P, 1, 0, 1, 1, V, 0);
      resPtr = xresf;
    } else resPtr = xf;

    // outs: (1,5) conv over V, K = 3*ci*5, fused BN+res+lrelu
    stt_conv_gemm<<<dim3(P / 128, co / 16), 128, 0, stream>>>(
        bp.outs_w, xsb, bp.outs_b, bp.outs_bn.g, bp.outs_bn.b, bp.outs_bn.m, bp.outs_bn.v,
        resPtr, y1f, y1b, co, Hh * ci, P, 5, 2, 1, 1, V, 1);

    // ff: 1x1 conv, fused BN+res+lrelu
    stt_conv_gemm<<<dim3(P / 128, co / 16), 128, 0, stream>>>(
        bp.ff_w, y1b, bp.ff_b, bp.ff_bn.g, bp.ff_bn.b, bp.ff_bn.m, bp.ff_bn.v,
        resPtr, y2f, y2b, co, co, P, 1, 0, 1, 1, V, 1);

    // temporal residual
    const float* restPtr;
    if (ci != co) {
      stt_conv_gemm<<<dim3(P / 128, co / 16), 128, 0, stream>>>(
          bp.rest_w, y2b, bp.rest_b, bp.rest_bn.g, bp.rest_bn.b, bp.rest_bn.m, bp.rest_bn.v,
          nullptr, restf, nullptr, co, co, P, 1, 0, 1, 1, V, 0);
      restPtr = restf;
    } else restPtr = y2f;

    // outt: (3,1) conv over T, fused BN+res+lrelu -> next block input (f32+bf16)
    stt_conv_gemm<<<dim3(P / 128, co / 16), 128, 0, stream>>>(
        bp.outt_w, y2b, bp.outt_b, bp.outt_bn.g, bp.outt_bn.b, bp.outt_bn.m, bp.outt_bn.v,
        restPtr, xf, xb, co, co, P, 3, 1, V, V, T, 1);
  }

  // ---- head: mean over (T,V), fold the two input halves, FC to 60 classes ----
  stt_pool<<<(Nn * Cmax) / 4, 128, 0, stream>>>(xf, poolf, Cmax, P, T * V);
  stt_fc<<<(32 * 60 + 127) / 128, 128, 0, stream>>>(poolf, fc_w, fc_b,
                                                    (float*)d_out, Cmax, 32, 60);
}
